// CG_model_40355512713748
// MI455X (gfx1250) — matrix operand order repaired and checked by reading, not compiled, not run on hardware
//
#include <hip/hip_runtime.h>
#include <hip/hip_bf16.h>

typedef __attribute__((ext_vector_type(16))) _Float16 v16h;
typedef __attribute__((ext_vector_type(8)))  _Float16 v8h;
typedef __attribute__((ext_vector_type(8)))  float    v8f;

#define NN   100000
#define EE   1000000
#define HCUT 4.0f
#define BOXL 4.0f
#define WPB  8          // waves per block (256 threads, wave32)

struct MLP6 { const float *W0, *b0, *W1, *b1, *W2, *b2; };

__device__ __forceinline__ float frcp(float x) { return __builtin_amdgcn_rcpf(x); }
__device__ __forceinline__ float fsig(float z) { return frcp(1.0f + __expf(-z)); }

// -------- WMMA helpers (gfx1250 wave32, 16x16x32 f16 -> f32) --------
__device__ __forceinline__ v8f wmma32f16(v16h a, v16h b, v8f c) {
  return __builtin_amdgcn_wmma_f32_16x16x32_f16(false, a, false, b, (short)0, c, false, false);
}

// A fragment from an f16 LDS tile[16 rows][32 cols] (row m = item, col k = feature).
// ISA 16-bit A layout: lane L (g=L>>4, m=L&15): halves 0..7 = k in [8g,8g+8),
// halves 8..15 = k in [16+8g,16+8g+8)  -> two contiguous 16B LDS loads.
__device__ __forceinline__ v16h pack_A16h(const _Float16* tile, int lane) {
  int g = lane >> 4, m = lane & 15;
  v8h lo = *(const v8h*)(tile + m * 32 + 8 * g);
  v8h hi = *(const v8h*)(tile + m * 32 + 16 + 8 * g);
  return __builtin_shufflevector(lo, hi, 0, 1, 2, 3, 4, 5, 6, 7, 8, 9, 10, 11, 12, 13, 14, 15);
}

// B fragment: B = W^T tile (32 x 16 cols [n0,n0+16)), W stored [nout][32] f32 in LDS.
__device__ __forceinline__ v16h pack_B16(const float* W, int n0, int nout, int lane) {
  int n = n0 + (lane & 15);
  int ncl = (n < nout) ? n : 0;
  int kb = 8 * (lane >> 4);
  v16h B;
#pragma unroll
  for (int h = 0; h < 16; ++h) {
    float w = W[ncl * 32 + kb + (h & 7) + ((h >> 3) << 4)];
    B[h] = (_Float16)((n < nout) ? w : 0.0f);
  }
  return B;
}

// D fragment (f32 16x16) -> f16 LDS tile, column block n0.
__device__ __forceinline__ void store_D16h(_Float16* tile, v8f d, int n0, int lane) {
  int g = lane >> 4, n = lane & 15;
#pragma unroll
  for (int r = 0; r < 8; ++r) tile[(r + 8 * g) * 32 + n0 + n] = (_Float16)d[r];
}

// -------- 16-item-batched MLP forward + JVP (tangent wrt last input) --------
// in_dim==1: x=(rr), tangent d/d rr.  in_dim==2: x=(rr,tt), tangent d/d tt.
// silu activations, layers [in]->32->32->1.  WMMA for layers 2 and 3.
__device__ __forceinline__ void mlp16_fwd_jvp(
    const float* W0, const float* b0, int in_dim,
    const float* b1, const float* b2s,
    v16h B1lo, v16h B1hi, v16h B2,
    const float* rr, const float* tt,
    _Float16* tV, _Float16* tT, float* os,
    int lane, float& val, float& dval) {
  // layer 1 (scalar): each lane owns output column o = lane
  float w0a, w0b, b0o = b0[lane];
  if (in_dim == 1) { w0a = W0[lane]; w0b = 0.0f; }
  else             { w0a = W0[lane * 2]; w0b = W0[lane * 2 + 1]; }
  float dz = (in_dim == 1) ? w0a : w0b;
#pragma unroll
  for (int m = 0; m < 16; ++m) {
    float z = (in_dim == 1) ? (w0a * rr[m] + b0o)
                            : (w0a * rr[m] + w0b * tt[m] + b0o);
    float sg = fsig(z);
    int idx = m * 32 + lane;
    tV[idx] = (_Float16)(z * sg);                               // silu
    tT[idx] = (_Float16)(sg * (1.0f + z * (1.0f - sg)) * dz);   // silu' * dz
  }
  // layer 2 via WMMA
  v16h Av = pack_A16h(tV, lane);
  v16h At = pack_A16h(tT, lane);
  v8f zero = {};
  v8f zlo = wmma32f16(Av, B1lo, zero);
  v8f zhi = wmma32f16(Av, B1hi, zero);
  v8f tlo = wmma32f16(At, B1lo, zero);
  v8f thi = wmma32f16(At, B1hi, zero);
  int n = lane & 15;
  float blo = b1[n], bhi = b1[16 + n];
  v8f alo, ahi, dlo, dhi;
#pragma unroll
  for (int r = 0; r < 8; ++r) {
    float z = zlo[r] + blo; float sg = fsig(z);
    alo[r] = z * sg; dlo[r] = sg * (1.0f + z * (1.0f - sg)) * tlo[r];
    z = zhi[r] + bhi; sg = fsig(z);
    ahi[r] = z * sg; dhi[r] = sg * (1.0f + z * (1.0f - sg)) * thi[r];
  }
  store_D16h(tV, alo, 0, lane);  store_D16h(tV, ahi, 16, lane);
  store_D16h(tT, dlo, 0, lane);  store_D16h(tT, dhi, 16, lane);
  // layer 3 via WMMA (32 -> 1): only column 0 of B live
  v8f ov = wmma32f16(pack_A16h(tV, lane), B2, zero);
  v8f ot = wmma32f16(pack_A16h(tT, lane), B2, zero);
  if ((lane & 15) == 0) {
    int g = lane >> 4;
#pragma unroll
    for (int r = 0; r < 8; ++r) { os[r + 8 * g] = ov[r] + b2s[0]; os[16 + r + 8 * g] = ot[r]; }
  }
  val  = os[lane & 15];
  dval = os[16 + (lane & 15)];
}

// -------- k0: zero accumulators + self-term kernel_w(0) --------
__global__ __launch_bounds__(256) void k0_init(float* ws, MLP6 Wn) {
  long idx = (long)blockIdx.x * 256 + threadIdx.x;
  if (idx < 8L * NN) ws[idx] = 0.0f;
  if (blockIdx.x == 0 && threadIdx.x == 0) {
    float a1[32];
#pragma unroll 4
    for (int o = 0; o < 32; ++o) { float z = Wn.b0[o]; a1[o] = z * fsig(z); }
    float out = Wn.b2[0];
    for (int o = 0; o < 32; ++o) {
      float z = Wn.b1[o];
      for (int k = 0; k < 32; ++k) z += Wn.W1[o * 32 + k] * a1[k];
      out += Wn.W2[o] * (z * fsig(z));
    }
    ws[8L * NN + (long)EE] = __expf(out);  // * (1 - 0^2)
  }
}

// -------- k1: W-net per edge (WMMA), density atomics, store dW/ds --------
__global__ __launch_bounds__(256) void k1_wnet(
    const float* __restrict__ x, const int* __restrict__ eidx,
    MLP6 Wn, float* dens, float* dWe) {
  __shared__ float sW1[1024], sW0[32], sb0[32], sb1[32], sW2[32], sb2[1];
  __shared__ _Float16 hst[WPB][1024];
  __shared__ float fst[WPB][64];
  int tid = threadIdx.x;
  for (int q = tid; q < 1024; q += 256) sW1[q] = Wn.W1[q];
  if (tid < 32) { sW0[tid] = Wn.W0[tid]; sb0[tid] = Wn.b0[tid]; sb1[tid] = Wn.b1[tid]; sW2[tid] = Wn.W2[tid]; }
  if (tid == 0) sb2[0] = Wn.b2[0];
  __syncthreads();
  int lane = tid & 31, wid = tid >> 5;
  v16h B1lo = pack_B16(sW1, 0, 32, lane);
  v16h B1hi = pack_B16(sW1, 16, 32, lane);
  v16h B2   = pack_B16(sW2, 0, 1, lane);
  _Float16* tV = hst[wid]; _Float16* tT = tV + 512;
  float* rr = fst[wid]; float* tt = rr + 16; float* os = rr + 32;
  int ntiles = (EE + 15) / 16;
  for (int t = blockIdx.x * WPB + wid; t < ntiles; t += gridDim.x * WPB) {
    int e = t * 16 + (lane & 15);
    bool valid = (lane < 16) && (e < EE);
    float s = 0.0f; int i = 0, jn = 0;
    if (valid) {
      i = eidx[e]; jn = eidx[EE + e];
      float dx = x[i * 7 + 0] - x[jn * 7 + 0]; dx -= BOXL * rintf(dx * (1.0f / BOXL));
      float dy = x[i * 7 + 1] - x[jn * 7 + 1]; dy -= BOXL * rintf(dy * (1.0f / BOXL));
      float dz = x[i * 7 + 2] - x[jn * 7 + 2]; dz -= BOXL * rintf(dz * (1.0f / BOXL));
      s = sqrtf(dx * dx + dy * dy + dz * dz) * (1.0f / HCUT);
    }
    if (lane < 16) rr[lane] = valid ? s : 0.0f;
    float mv, mt;
    mlp16_fwd_jvp(sW0, sb0, 1, sb1, sb2, B1lo, B1hi, B2, rr, tt, tV, tT, os, lane, mv, mt);
    if (valid) {
      float em = __expf(mv);
      float w  = em * (1.0f - s * s);
      dWe[e]   = em * (mt * (1.0f - s * s) - 2.0f * s);  // dW/ds
      atomicAdd(&dens[i], w);
      atomicAdd(&dens[jn], w);
    }
  }
}

// -------- k2: per-node monotone U-net via WMMA batches of 16 nodes --------
// Produces T = dU/dS, Cc = T / d2U/dS2, gd = dU/d(density).
__global__ __launch_bounds__(256) void k2_node(
    const float* __restrict__ x, MLP6 Un,
    const float* dens, const float* w0self,
    float* Tn, float* Ccn, float* gdn) {
  __shared__ float uW0[64], ub0[32], uW1[1024], ub1[32], uW2[32];
  __shared__ _Float16 hst[WPB][2048];   // 4 f16 tiles of 16x32
  __shared__ float fst[WPB][80];        // S[16] V[16] osT[16] osQ[16] osU[16]
  int tid = threadIdx.x;
  for (int q = tid; q < 1024; q += 256) uW1[q] = fabsf(Un.W1[q]);
  if (tid < 64) uW0[tid] = (tid & 1) ? -fabsf(Un.W0[tid]) : fabsf(Un.W0[tid]);  // t=[1,-1]
  if (tid < 32) { ub0[tid] = Un.b0[tid]; ub1[tid] = Un.b1[tid]; uW2[tid] = fabsf(Un.W2[tid]); }
  __syncthreads();
  int lane = tid & 31, wid = tid >> 5;
  v16h B1lo = pack_B16(uW1, 0, 32, lane);
  v16h B1hi = pack_B16(uW1, 16, 32, lane);
  v16h B2u  = pack_B16(uW2, 0, 1, lane);
  _Float16* hA  = hst[wid];        // a1 (softplus)   -> later p tile
  _Float16* hS  = hA + 512;        // d a1/ds         -> later q tile
  _Float16* hSS = hS + 512;        // d2 a1/ds2       -> later u tile
  _Float16* hV  = hSS + 512;       // d a1/dV
  float* sS = fst[wid]; float* sV = sS + 16;
  float* osT = sV + 16; float* osQ = osT + 16; float* osU = osQ + 16;
  float w0 = w0self[0];
  float w0a = uW0[lane * 2], w0b = uW0[lane * 2 + 1], b0o = ub0[lane];  // column o = lane
  int ntiles = (NN + 15) / 16;
  for (int t = blockIdx.x * WPB + wid; t < ntiles; t += gridDim.x * WPB) {
    int n = t * 16 + (lane & 15);
    bool valid = (lane < 16) && (n < NN);
    float Vv = 1.0f, Sv = 0.0f;
    if (valid) { Vv = frcp(dens[n] + w0); Sv = x[n * 7 + 6]; }
    if (lane < 16) { sS[lane] = valid ? Sv : 0.0f; sV[lane] = valid ? Vv : 1.0f; }
    // layer 1 (scalar, column o = lane): value + 3 tangent chains
#pragma unroll
    for (int m = 0; m < 16; ++m) {
      float z = w0a * sS[m] + w0b * sV[m] + b0o;
      float sg = fsig(z);
      float sp = (z > 20.0f) ? z : log1pf(__expf(z));   // softplus
      int idx = m * 32 + lane;
      hA[idx]  = (_Float16)sp;
      hS[idx]  = (_Float16)(sg * w0a);
      hSS[idx] = (_Float16)(sg * (1.0f - sg) * w0a * w0a);
      hV[idx]  = (_Float16)(sg * w0b);
    }
    v16h Aa  = pack_A16h(hA, lane);
    v16h As  = pack_A16h(hS, lane);
    v16h Ass = pack_A16h(hSS, lane);
    v16h Avv = pack_A16h(hV, lane);
    v8f zero = {};
    v8f z2lo = wmma32f16(Aa, B1lo, zero),  z2hi = wmma32f16(Aa, B1hi, zero);
    v8f s2lo = wmma32f16(As, B1lo, zero),  s2hi = wmma32f16(As, B1hi, zero);
    v8f q2lo = wmma32f16(Ass, B1lo, zero), q2hi = wmma32f16(Ass, B1hi, zero);
    v8f v2lo = wmma32f16(Avv, B1lo, zero), v2hi = wmma32f16(Avv, B1hi, zero);
    int nn = lane & 15;
    float blo = ub1[nn], bhi = ub1[16 + nn];
    v8f plo, phi, qlo, qhi, ulo, uhi;
#pragma unroll
    for (int r = 0; r < 8; ++r) {
      float z = z2lo[r] + blo; float sg = fsig(z); float sd = sg * (1.0f - sg);
      plo[r] = sg * s2lo[r];
      qlo[r] = sd * s2lo[r] * s2lo[r] + sg * q2lo[r];
      ulo[r] = sg * v2lo[r];
      z = z2hi[r] + bhi; sg = fsig(z); sd = sg * (1.0f - sg);
      phi[r] = sg * s2hi[r];
      qhi[r] = sd * s2hi[r] * s2hi[r] + sg * q2hi[r];
      uhi[r] = sg * v2hi[r];
    }
    store_D16h(hA, plo, 0, lane);  store_D16h(hA, phi, 16, lane);
    store_D16h(hS, qlo, 0, lane);  store_D16h(hS, qhi, 16, lane);
    store_D16h(hSS, ulo, 0, lane); store_D16h(hSS, uhi, 16, lane);
    // w2-weighted feature reductions via WMMA (col 0 of B2u)
    v8f Tm = wmma32f16(pack_A16h(hA, lane), B2u, zero);
    v8f Qm = wmma32f16(pack_A16h(hS, lane), B2u, zero);
    v8f Um = wmma32f16(pack_A16h(hSS, lane), B2u, zero);
    if ((lane & 15) == 0) {
      int g = lane >> 4;
#pragma unroll
      for (int r = 0; r < 8; ++r) { osT[r + 8 * g] = Tm[r]; osQ[r + 8 * g] = Qm[r]; osU[r + 8 * g] = Um[r]; }
    }
    if (valid) {
      float T = osT[lane], Tpp = osQ[lane], UV = osU[lane];
      Tn[n] = T;
      Ccn[n] = T * frcp(Tpp);
      gdn[n] = -UV * Vv * Vv;     // dU/d(density)
    }
  }
}

// -------- k3: A/B/C edge nets (WMMA) + physics + fused atomics --------
__global__ __launch_bounds__(256) void k3_edge(
    const float* __restrict__ x, const int* __restrict__ eidx,
    MLP6 An, MLP6 Bn, MLP6 Cn,
    const float* Tn, const float* Ccn, const float* gdn,
    const float* dWe, const float* lkB, const float* lm,
    float* accV, float* accS) {
  __shared__ float sW0[3][64], sb0[3][32], sW1[3][1024], sb1[3][32], sW2[3][32], sb2[3][1];
  __shared__ _Float16 hst[WPB][1024];
  __shared__ float fst[WPB][64];
  int tid = threadIdx.x;
  const MLP6 nets[3] = {An, Bn, Cn};
  for (int q = 0; q < 3; ++q) {
    for (int idx = tid; idx < 1024; idx += 256) sW1[q][idx] = nets[q].W1[idx];
    if (tid < 64) sW0[q][tid] = nets[q].W0[tid];
    if (tid < 32) { sb0[q][tid] = nets[q].b0[tid]; sb1[q][tid] = nets[q].b1[tid]; sW2[q][tid] = nets[q].W2[tid]; }
    if (tid == 0) sb2[q][0] = nets[q].b2[0];
  }
  __syncthreads();
  int lane = tid & 31, wid = tid >> 5;
  v16h B1lo[3], B1hi[3], B2f[3];
#pragma unroll
  for (int q = 0; q < 3; ++q) {
    B1lo[q] = pack_B16(sW1[q], 0, 32, lane);
    B1hi[q] = pack_B16(sW1[q], 16, 32, lane);
    B2f[q]  = pack_B16(sW2[q], 0, 1, lane);
  }
  float kB = __expf(lkB[0]);
  float inv_m = __expf(-lm[0]);
  _Float16* tV = hst[wid]; _Float16* tT = tV + 512;
  float* rr = fst[wid]; float* tt = rr + 16; float* os = rr + 32;
  const float invD = 1.0f / 3.0f;
  int ntiles = (EE + 15) / 16;
  for (int t = blockIdx.x * WPB + wid; t < ntiles; t += gridDim.x * WPB) {
    int e = t * 16 + (lane & 15);
    bool valid = (lane < 16) && (e < EE);
    int i = 0, jn = 0;
    float eij[3] = {0, 0, 0}, vij[3] = {0, 0, 0}, ev = 0, vv = 0, srr = 0;
    float Ti = 1, Tj = 1, Ci = 1, Cj = 1, gdi = 0, gdj = 0, dW = 0;
    if (valid) {
      i = eidx[e]; jn = eidx[EE + e];
      float rv[3];
#pragma unroll
      for (int d = 0; d < 3; ++d) {
        float dr = x[i * 7 + d] - x[jn * 7 + d];
        dr -= BOXL * rintf(dr * (1.0f / BOXL));
        rv[d] = dr;
        vij[d] = x[i * 7 + 3 + d] - x[jn * 7 + 3 + d];
      }
      float rn = sqrtf(rv[0] * rv[0] + rv[1] * rv[1] + rv[2] * rv[2]);
      float inv = frcp(rn + 1e-8f);
#pragma unroll
      for (int d = 0; d < 3; ++d) { eij[d] = rv[d] * inv; ev += eij[d] * vij[d]; vv += vij[d] * vij[d]; }
      srr = rn * (1.0f / HCUT);
      Ti = Tn[i]; Tj = Tn[jn]; Ci = Ccn[i]; Cj = Ccn[jn]; gdi = gdn[i]; gdj = gdn[jn]; dW = dWe[e];
    }
    if (lane < 16) rr[lane] = valid ? srr : 0.0f;
    float av[2][3], ad[2][3];
#pragma unroll
    for (int side = 0; side < 2; ++side) {
      if (lane < 16) tt[lane] = valid ? (side ? Tj : Ti) : 0.0f;
#pragma unroll
      for (int q = 0; q < 3; ++q) {
        float v_, d_;
        mlp16_fwd_jvp(sW0[q], sb0[q], 2, sb1[q], sb2[q],
                      B1lo[q], B1hi[q], B2f[q], rr, tt, tV, tT, os, lane, v_, d_);
        av[side][q] = v_; ad[side][q] = d_;
      }
    }
    if (valid) {
      float a_i = av[0][0], a_j = av[1][0], da_i = ad[0][0], da_j = ad[1][0];
      float b_i = av[0][1], b_j = av[1][1], db_i = ad[0][1], db_j = ad[1][1];
      float c_i = av[0][2], c_j = av[1][2], dc_i = ad[0][2], dc_j = ad[1][2];
      float A  = a_i * a_j, Bv = b_i * b_j, Cij = c_i * c_j;
      float gA_i = 2.0f * A * a_j * da_i, gA_j = 2.0f * A * a_i * da_j;
      float gB_i = 2.0f * Bv * b_j * db_i, gB_j = 2.0f * Bv * b_i * db_j;
      float gC_i = 2.0f * Cij * c_j * dc_i, gC_j = 2.0f * Cij * c_i * dc_j;
      float A2 = A * A, B2 = Bv * Bv, C2 = Cij * Cij;
      float coef = 0.5f * A2 + (B2 - A2) * invD;
      float invTi = frcp(Ti), invTj = frcp(Tj);
      float rCi = frcp(Ci), rCj = frcp(Cj);
      float iCTi = rCi * invTi, iCTj = rCj * invTj;
      float term2c = -(iCTi + iCTj);
      float ci_c = 0.5f * gA_i + (gB_i - gA_i) * invD;
      float cj_c = 0.5f * gA_j + (gB_j - gA_j) * invD;
      float fF = -(gdi + gdj) * dW * (1.0f / HCUT);     // F_cons edge factor
      float P[3];
#pragma unroll
      for (int d = 0; d < 3; ++d) {
        float aux   = 0.5f * A2 * vij[d] + coef * ev * eij[d];
        float vec_i = (0.5f * gA_i * vij[d] + ci_c * ev * eij[d]) * rCi;
        float vec_j = (0.5f * gA_j * vij[d] + cj_c * ev * eij[d]) * rCj;
        float w = term2c * aux + vec_i + vec_j;
        // fused: F_cons + MgradS_v + kB*divM_v contribution at node i (negated at j)
        P[d] = fF * eij[d] - 0.5f * (invTi + invTj) * aux - 0.5f * kB * w;
      }
      float aux_s = 0.25f * (0.5f * A2 * vv + coef * ev * ev);
      float Msi = (invTi + invTj) * aux_s + (invTi - invTj) * C2;
      float Msj = (invTi + invTj) * aux_s + (invTj - invTi) * C2;
      float t1i = -(2.0f * iCTi + iCTj) * aux_s, t1j = -(2.0f * iCTj + iCTi) * aux_s;
      float w_i  = (0.5f * gA_i * vv + (0.5f * gA_i + (gB_i - gA_i) * invD) * ev * ev) * rCi * 0.25f;
      float w_j  = (0.5f * gA_j * vv + (0.5f * gA_j + (gB_j - gA_j) * invD) * ev * ev) * rCj * 0.25f;
      float w3j  = (0.5f * gA_i * vv + (0.5f * gA_j + (gB_i - gA_i) * invD) * ev * ev) * rCi * 0.25f;
      float t4i = -(2.0f * iCTi - iCTj) * C2, t4j = -(2.0f * iCTj - iCTi) * C2;
      float t5  = gC_i * rCi - gC_j * rCj;
      float tc  = -(2.0f * A2 + (B2 - A2) * invD);      // (D+1)/2 = 2 for D=3
      float Si = Msi + kB * (t1i + w_i + w_j + t4i + t5 + tc * inv_m);
      float Sj = Msj + kB * (t1j + w_j + w3j + t4j - t5 + tc * inv_m);
      atomicAdd(&accV[i * 3 + 0],  P[0]);  atomicAdd(&accV[i * 3 + 1],  P[1]);  atomicAdd(&accV[i * 3 + 2],  P[2]);
      atomicAdd(&accV[jn * 3 + 0], -P[0]); atomicAdd(&accV[jn * 3 + 1], -P[1]); atomicAdd(&accV[jn * 3 + 2], -P[2]);
      atomicAdd(&accS[i], Si);
      atomicAdd(&accS[jn], Sj);
    }
  }
}

// -------- k4: finalize outputs --------
__global__ __launch_bounds__(256) void k4_out(
    const float* __restrict__ x, const float* accV, const float* accS,
    const float* Tn, const float* lm, float* out) {
  int n = blockIdx.x * 256 + threadIdx.x;
  if (n >= NN) return;
  float inv_m = __expf(-lm[0]);
#pragma unroll
  for (int d = 0; d < 3; ++d) {
    out[n * 3 + d] = x[n * 7 + 3 + d];                       // drdt = v
    out[3 * NN + n * 3 + d] = accV[n * 3 + d] * inv_m;       // dvdt
  }
  out[6 * NN + n] = accS[n] * frcp(Tn[n]);                   // dSdt
}

extern "C" void kernel_launch(void* const* d_in, const int* in_sizes, int n_in,
                              void* d_out, int out_size, void* d_ws, size_t ws_size,
                              hipStream_t stream) {
  (void)in_sizes; (void)n_in; (void)out_size; (void)ws_size;
  const float* x    = (const float*)d_in[0];
  const int*   eidx = (const int*)d_in[1];
  auto mk = [&](int b) {
    MLP6 m;
    m.W0 = (const float*)d_in[b + 0]; m.b0 = (const float*)d_in[b + 1];
    m.W1 = (const float*)d_in[b + 2]; m.b1 = (const float*)d_in[b + 3];
    m.W2 = (const float*)d_in[b + 4]; m.b2 = (const float*)d_in[b + 5];
    return m;
  };
  MLP6 Wn = mk(3), Un = mk(9), An = mk(15), Bn = mk(21), Cn = mk(27);
  const float* lkB = (const float*)d_in[33];
  const float* lm  = (const float*)d_in[34];

  float* ws     = (float*)d_ws;
  float* dens   = ws;            // N
  float* Tnp    = ws + 1L * NN;  // N
  float* Ccn    = ws + 2L * NN;  // N
  float* gdn    = ws + 3L * NN;  // N
  float* accS   = ws + 4L * NN;  // N
  float* accV   = ws + 5L * NN;  // 3N
  float* dWe    = ws + 8L * NN;  // E
  float* w0self = ws + 8L * NN + (long)EE;  // 1
  float* out    = (float*)d_out;

  int zb = (8 * NN + 255) / 256;
  int etiles = (EE + 15) / 16;
  int eb = (etiles + WPB - 1) / WPB;
  int ntl = (NN + 15) / 16;
  int n2b = (ntl + WPB - 1) / WPB;
  int nb = (NN + 255) / 256;

  k0_init<<<zb, 256, 0, stream>>>(ws, Wn);
  k1_wnet<<<eb, 256, 0, stream>>>(x, eidx, Wn, dens, dWe);
  k2_node<<<n2b, 256, 0, stream>>>(x, Un, dens, w0self, Tnp, Ccn, gdn);
  k3_edge<<<eb, 256, 0, stream>>>(x, eidx, An, Bn, Cn, Tnp, Ccn, gdn, dWe, lkB, lm, accV, accS);
  k4_out<<<nb, 256, 0, stream>>>(x, accV, accS, Tnp, lm, out);
}
